// BahdanauAttention_40647570489776
// MI455X (gfx1250) — matrix operand order repaired and checked by reading, not compiled, not run on hardware
//
#include <hip/hip_runtime.h>
#include <hip/hip_bf16.h>

typedef _Float16 v16h __attribute__((ext_vector_type(16)));
typedef _Float16 v8h  __attribute__((ext_vector_type(8)));
typedef float    v8f  __attribute__((ext_vector_type(8)));

#define NB    8
#define LQ    2048
#define LK    2048
#define DDIM  256
#define BM    64
#define BK    32

// ---------------- Kernel 1: sk[b,j] = key[b,j,:] . w  (wave per row) ----------------
__global__ __launch_bounds__(256)
void sk_dot_kernel(const float* __restrict__ key, const float* __restrict__ w,
                   float* __restrict__ sk) {
  const int wave = threadIdx.x >> 5;
  const int lane = threadIdx.x & 31;
  const int row  = blockIdx.x * 8 + wave;            // row in [0, NB*LK)
  const float* kp = key + (size_t)row * DDIM;
  float s = 0.f;
#pragma unroll
  for (int i = 0; i < DDIM; i += 32) s += kp[i + lane] * w[i + lane];
#pragma unroll
  for (int off = 16; off > 0; off >>= 1) s += __shfl_down(s, off, 32);
  if (lane == 0) sk[row] = s;
}

// ---------------- Kernel 2: per-batch max, e = exp(sk - m), in place ----------------
__global__ __launch_bounds__(256)
void softmax_exp_kernel(float* __restrict__ sk) {
  float* p = sk + (size_t)blockIdx.x * LK;
  __shared__ float red[256];
  float m = -3.0e38f;
  for (int j = threadIdx.x; j < LK; j += 256) m = fmaxf(m, p[j]);
  red[threadIdx.x] = m;
  __syncthreads();
  for (int s = 128; s > 0; s >>= 1) {
    if (threadIdx.x < s) red[threadIdx.x] = fmaxf(red[threadIdx.x], red[threadIdx.x + s]);
    __syncthreads();
  }
  m = red[0];
  for (int j = threadIdx.x; j < LK; j += 256) p[j] = expf(p[j] - m);
}

// ---------------- Kernel 3: out = (M x (e*V)) / (M x e), WMMA f16 ----------------
__global__ __launch_bounds__(256)
void attn_gemm_kernel(const int* __restrict__ mask,
                      const float* __restrict__ value,
                      const float* __restrict__ e_g,
                      float* __restrict__ out) {
  __shared__ alignas(16) _Float16 As[BM * BK];     // [row][k], stride 32 halves
  __shared__ alignas(16) _Float16 Bs[DDIM * BK];   // [col][k] (transposed), stride 32
  __shared__ float e_s[BK];
  __shared__ float den_part[256];
  __shared__ float den_s[BM];

  const int mb  = blockIdx.x & 31;                 // LQ/BM = 32 tiles
  const int b   = blockIdx.x >> 5;
  const int m0  = mb * BM;
  const int tid = threadIdx.x;
  const int lane = tid & 31, wave = tid >> 5;
  const int m_off  = (wave & 3) * 16;              // 4 M-frags
  const int n_base = (wave >> 2) * 128;            // 2 N-halves
  const int h = lane >> 4, n = lane & 15;

  v8f acc[8];
#pragma unroll
  for (int f = 0; f < 8; ++f) acc[f] = {};
  float denp = 0.f;

  // loader assignments
  const int a_row = tid >> 2, a_col = (tid & 3) * 8;      // A: 64x32 ints, 8/thread
  const int b_k   = tid >> 3, b_d  = (tid & 7) * 32;      // B: 32x256 floats, 32/thread
  const int*   mrow = mask  + ((size_t)(b * LQ + m0 + a_row)) * LK + a_col;
  const float* vrow = value + ((size_t)(b * LK + b_k)) * DDIM + b_d;
  const float* e_b  = e_g + (size_t)b * LK;

  for (int kk = 0; kk < LK; kk += BK) {
    __syncthreads();
    // --- stage A: mask tile -> exact 0/1 f16 ---
    {
      const int* mp = mrow + kk;
      __builtin_prefetch(mp + BK, 0, 1);          // next tile -> global_prefetch_b8
      int4 q0 = *(const int4*)(mp);
      int4 q1 = *(const int4*)(mp + 4);
      _Float16* ap = As + a_row * BK + a_col;
      ap[0] = q0.x ? (_Float16)1.f : (_Float16)0.f;
      ap[1] = q0.y ? (_Float16)1.f : (_Float16)0.f;
      ap[2] = q0.z ? (_Float16)1.f : (_Float16)0.f;
      ap[3] = q0.w ? (_Float16)1.f : (_Float16)0.f;
      ap[4] = q1.x ? (_Float16)1.f : (_Float16)0.f;
      ap[5] = q1.y ? (_Float16)1.f : (_Float16)0.f;
      ap[6] = q1.z ? (_Float16)1.f : (_Float16)0.f;
      ap[7] = q1.w ? (_Float16)1.f : (_Float16)0.f;
    }
    if (tid < BK) e_s[tid] = e_b[kk + tid];
    // --- stage B: (e_j * v[j,d]) -> f16, transposed [col][k] ---
    {
      const float ev = e_b[kk + b_k];
      const float* vp = vrow + (size_t)kk * DDIM;
#pragma unroll
      for (int i = 0; i < 32; i += 4) {
        float4 v4 = *(const float4*)(vp + i);
        Bs[(b_d + i + 0) * BK + b_k] = (_Float16)(v4.x * ev);
        Bs[(b_d + i + 1) * BK + b_k] = (_Float16)(v4.y * ev);
        Bs[(b_d + i + 2) * BK + b_k] = (_Float16)(v4.z * ev);
        Bs[(b_d + i + 3) * BK + b_k] = (_Float16)(v4.w * ev);
      }
    }
    __syncthreads();

    // --- den partial: row a_row, ks [a_col, a_col+8) ---
    {
      const _Float16* ar = As + a_row * BK + a_col;
#pragma unroll
      for (int i = 0; i < 8; ++i) denp += (float)ar[i] * e_s[a_col + i];
    }

    // --- WMMA: A frag per wave, 8 B frags ---
    // A 16x32 f16 layout: lane = m + 16h; VGPR r<4: K = 8h+2r(,+1); r>=4: K = 16+8h+2(r-4)
    const _Float16* arow = As + (m_off + n) * BK;
    v8h alo = *(const v8h*)(arow + 8 * h);         // K = 8h .. 8h+7
    v8h ahi = *(const v8h*)(arow + 16 + 8 * h);    // K = 16+8h .. 16+8h+7
    v16h afrag = __builtin_shufflevector(alo, ahi, 0,1,2,3,4,5,6,7,8,9,10,11,12,13,14,15);
#pragma unroll
    for (int f = 0; f < 8; ++f) {
      // B 32x16 f16 layout: lane = n + 16h; VGPR r<4: K = 8h+2r(,+1); r>=4: K = 16+8h+2(r-4)
      const _Float16* bcol = Bs + (n_base + f * 16 + n) * BK;
      v8h blo = *(const v8h*)(bcol + 8 * h);       // K = 8h .. 8h+7
      v8h bhi = *(const v8h*)(bcol + 16 + 8 * h);  // K = 16+8h .. 16+8h+7
      v16h bfrag = __builtin_shufflevector(blo, bhi, 0,1,2,3,4,5,6,7,8,9,10,11,12,13,14,15);
      acc[f] = __builtin_amdgcn_wmma_f32_16x16x32_f16(
          false, afrag, false, bfrag, (short)0, acc[f], false, false);
    }
  }

  // --- reduce den: 4 partials per row ---
  den_part[tid] = denp;
  __syncthreads();
  if (tid < BM)
    den_s[tid] = den_part[tid * 4] + den_part[tid * 4 + 1] +
                 den_part[tid * 4 + 2] + den_part[tid * 4 + 3];
  __syncthreads();

  // --- epilogue: C layout VGPR v: (M = v + 8h, N = n); divide by den ---
  float dr[8];
#pragma unroll
  for (int v = 0; v < 8; ++v) dr[v] = den_s[m_off + 8 * h + v];
  float* op = out + ((size_t)b * LQ + m0) * DDIM;
#pragma unroll
  for (int f = 0; f < 8; ++f) {
    const int col = n_base + f * 16 + n;
#pragma unroll
    for (int v = 0; v < 8; ++v) {
      const int rowl = m_off + 8 * h + v;
      op[(size_t)rowl * DDIM + col] = acc[f][v] / dr[v];
    }
  }
}

extern "C" void kernel_launch(void* const* d_in, const int* in_sizes, int n_in,
                              void* d_out, int out_size, void* d_ws, size_t ws_size,
                              hipStream_t stream) {
  // inputs: query(unused — cancels in softmax), key, value, mask(int32), w_align
  const float* key   = (const float*)d_in[1];
  const float* value = (const float*)d_in[2];
  const int*   mask  = (const int*)d_in[3];
  const float* w     = (const float*)d_in[4];
  float* out = (float*)d_out;
  float* sk  = (float*)d_ws;                      // NB*LK floats = 64 KB scratch

  sk_dot_kernel<<<(NB * LK) / 8, 256, 0, stream>>>(key, w, sk);
  softmax_exp_kernel<<<NB, 256, 0, stream>>>(sk);
  attn_gemm_kernel<<<NB * (LQ / BM), 256, 0, stream>>>(mask, value, sk, out);
}
